// GPT_11063835754591
// MI455X (gfx1250) — matrix operand order repaired and checked by reading, not compiled, not run on hardware
//
#include <hip/hip_runtime.h>

// ---------------------------------------------------------------------------
// GPT forward on gfx1250 (MI455X).  Compute-bound on GEMMs (~3.4 TFLOP vs
// ~0.5 GB of traffic at 23.3 TB/s), so all matmuls run through
// v_wmma_f32_16x16x32_bf16 with f32 accumulation.  Staging uses the CDNA5
// async-to-LDS path (global_load_async_to_lds_b128 + s_wait_asynccnt) with
// statically double-buffered, pair-unrolled software pipelining.
// ---------------------------------------------------------------------------

typedef unsigned short u16;
typedef unsigned int   u32;
typedef unsigned long long u64;

typedef __bf16 bf16x16 __attribute__((ext_vector_type(16)));
typedef float  f32x8   __attribute__((ext_vector_type(8)));

// Model constants
#define CB   8
#define CT   1024
#define CL   8
#define CH   8
#define CD   512
#define CV   8192
#define CHD  4096            // H*D
#define CFF  2048            // 4*D
#define CM   (CB * CT)       // 8192 token rows

__device__ __forceinline__ u16 f2bf(float f) {
  u32 u = __float_as_uint(f);
  u32 r = u + 0x7FFFu + ((u >> 16) & 1u);   // round-to-nearest-even
  return (u16)(r >> 16);
}

union FragU {
  bf16x16 v;
  struct { uint4 lo; uint4 hi; } q;
};

__device__ __forceinline__ void async_ld128(u32 lds_off, const u16* gptr) {
  // GLOBAL_LOAD_ASYNC_TO_LDS_B128: per-lane 16B  memory -> LDS, ASYNCcnt.
  asm volatile("global_load_async_to_lds_b128 %0, %1, off"
               :: "v"(lds_off), "v"((u64)(uintptr_t)gptr)
               : "memory");
}

// ---------------------------------------------------------------------------
// WMMA GEMM: C[z] = A[z] (M x K, row-major bf16, lda) * B[z]^T (N x K, ldb)
// Macro tile 128(M) x 128(N), 8 waves, each wave 32x64 via 2x4 WMMA tiles.
// Double-buffered async-to-LDS staging, K-step 32, pair-unrolled so LDS
// buffer addresses are compile-time static.
// Epilogue: + bias[col], GELU(exact), + residual, store bf16 or f32.
// ---------------------------------------------------------------------------
#define PITCH 40   // 32 + 8 pad (bf16 elems) -> 80B row pitch, conflict-free

template <bool OUTBF, bool BIAS, bool RES, bool GELU>
__global__ __launch_bounds__(256)
void gemm_bf16_wmma(const u16* __restrict__ A, long long aBatch, int lda,
                    const u16* __restrict__ Bm, long long bBatch, int ldb,
                    void* __restrict__ Cv, long long cBatch, int ldc,
                    const float* __restrict__ bias,
                    const float* __restrict__ resid, long long rBatch, int ldr,
                    int K) {
  __shared__ u16 sA[2][128 * PITCH];
  __shared__ u16 sB[2][128 * PITCH];

  const int tid  = threadIdx.x;
  const int lane = tid & 31;
  const int wv   = tid >> 5;          // 0..7
  const int wr   = wv & 3;            // wave row -> 4 * 32 = 128 rows
  const int wc   = wv >> 2;           // wave col -> 2 * 64 = 128 cols
  const int z    = blockIdx.z;

  const long long rowBase = (long long)blockIdx.y * 128;
  const long long colBase = (long long)blockIdx.x * 128;

  A  += (long long)z * aBatch;
  Bm += (long long)z * bBatch;

  // staging chunk coordinates: thread handles 16B chunks (r0,c0) and (r1,c0)
  // for both the A tile (128x32) and the B tile (128x32).
  const int r0 = tid >> 2,      c0 = tid & 3;
  const int r1 = 64 + (tid >> 2);

  const u16* gA0 = A  + (rowBase + r0) * lda + c0 * 8;
  const u16* gA1 = A  + (rowBase + r1) * lda + c0 * 8;
  const u16* gB0 = Bm + (colBase + r0) * ldb + c0 * 8;
  const u16* gB1 = Bm + (colBase + r1) * ldb + c0 * 8;

  u32 lA0[2], lA1[2], lB0[2], lB1[2];
#pragma unroll
  for (int bfi = 0; bfi < 2; ++bfi) {
    lA0[bfi] = (u32)(uintptr_t)&sA[bfi][r0 * PITCH + c0 * 8];
    lA1[bfi] = (u32)(uintptr_t)&sA[bfi][r1 * PITCH + c0 * 8];
    lB0[bfi] = (u32)(uintptr_t)&sB[bfi][r0 * PITCH + c0 * 8];
    lB1[bfi] = (u32)(uintptr_t)&sB[bfi][r1 * PITCH + c0 * 8];
  }

  f32x8 acc[2][4] = {};

  // issue one k-stage (4 async-LDS instructions per wave) into buffer `bfi`
  auto issue = [&](int bfi, int k0) {
    async_ld128(lA0[bfi], gA0 + k0);
    async_ld128(lA1[bfi], gA1 + k0);
    async_ld128(lB0[bfi], gB0 + k0);
    async_ld128(lB1[bfi], gB1 + k0);
  };

  // one 32-wide k-step of WMMA work from buffer `bfi` (static addresses)
  auto mma_step = [&](int bfi) {
    const int rr = lane & 15;
    const int kg = lane >> 4;
    const u16* bA = &sA[bfi][0];
    const u16* bB = &sB[bfi][0];
    FragU a[2], bfr[4];
#pragma unroll
    for (int t = 0; t < 2; ++t) {
      const u16* p = bA + (wr * 32 + t * 16 + rr) * PITCH + kg * 8;
      a[t].q.lo = *(const uint4*)(p);        // K = kg*8 .. +7
      a[t].q.hi = *(const uint4*)(p + 16);   // K = 16+kg*8 .. +7
    }
#pragma unroll
    for (int t = 0; t < 4; ++t) {
      const u16* p = bB + (wc * 64 + t * 16 + rr) * PITCH + kg * 8;
      bfr[t].q.lo = *(const uint4*)(p);      // K = kg*8 .. +7
      bfr[t].q.hi = *(const uint4*)(p + 16); // K = 16+kg*8 .. +7
    }
#pragma unroll
    for (int tr = 0; tr < 2; ++tr)
#pragma unroll
      for (int tc = 0; tc < 4; ++tc)
        acc[tr][tc] = __builtin_amdgcn_wmma_f32_16x16x32_bf16(
            false, a[tr].v, false, bfr[tc].v, (short)0, acc[tr][tc],
            false, false);
  };

  // ---- statically double-buffered pipeline (K/32 is even for all callers) --
  issue(0, 0);
  int k0 = 0;
  for (; k0 + 64 < K; k0 += 64) {
    issue(1, k0 + 32);
    asm volatile("s_wait_asynccnt 0x4" ::: "memory");  // stage in buf0 landed
    __syncthreads();
    mma_step(0);
    __syncthreads();                                   // buf0 free for reuse
    issue(0, k0 + 64);
    asm volatile("s_wait_asynccnt 0x4" ::: "memory");  // stage in buf1 landed
    __syncthreads();
    mma_step(1);
    __syncthreads();                                   // buf1 free for reuse
  }
  // final pair (k0, k0+32)
  issue(1, k0 + 32);
  asm volatile("s_wait_asynccnt 0x4" ::: "memory");
  __syncthreads();
  mma_step(0);
  __syncthreads();
  asm volatile("s_wait_asynccnt 0x0" ::: "memory");
  __syncthreads();
  mma_step(1);

  // ---- epilogue: C/D layout VGPR r -> M = r + 8*(lane>=16), N = lane&15 ----
  float* Cf = (float*)Cv;
  u16*   Ch = (u16*)Cv;
  const int cLane = lane & 15;
  const int rHalf = (lane >> 4) * 8;
#pragma unroll
  for (int tr = 0; tr < 2; ++tr) {
#pragma unroll
    for (int tc = 0; tc < 4; ++tc) {
      const long long col = colBase + wc * 64 + tc * 16 + cLane;
      float bv = 0.0f;
      if (BIAS) bv = bias[col];
#pragma unroll
      for (int r = 0; r < 8; ++r) {
        const long long row = rowBase + wr * 32 + tr * 16 + rHalf + r;
        float v = acc[tr][tc][r] + bv;
        if (GELU) v = 0.5f * v * (1.0f + erff(v * 0.70710678118654752f));
        if (RES)  v += resid[(long long)z * rBatch + row * (long long)ldr + col];
        const long long ci = (long long)z * cBatch + row * (long long)ldc + col;
        if (OUTBF) Ch[ci] = f2bf(v);
        else       Cf[ci] = v;
      }
    }
  }
}

// ---------------------------------------------------------------------------
// Elementwise / reduction helpers
// ---------------------------------------------------------------------------

__global__ __launch_bounds__(256)
void cvt_f32_bf16(const float* __restrict__ in, u16* __restrict__ out, int n) {
  int i = blockIdx.x * 256 + threadIdx.x;
  if (i < n) out[i] = f2bf(in[i]);
}

__global__ __launch_bounds__(256)
void embed_kernel(const int* __restrict__ idx, const float* __restrict__ tok,
                  const float* __restrict__ pos, float* __restrict__ x) {
  long long n = (long long)blockIdx.x * 256 + threadIdx.x;   // CM*CD total
  int  d  = (int)(n & (CD - 1));
  long long bt = n >> 9;            // CD = 512
  int  t  = (int)(bt & (CT - 1));
  x[n] = tok[(long long)idx[bt] * CD + d] + pos[(long long)t * CD + d];
}

// wave-per-row LayerNorm over D=512, bf16 output
__global__ __launch_bounds__(256)
void ln_rows(const float* __restrict__ x, const float* __restrict__ w,
             const float* __restrict__ b, u16* __restrict__ out) {
  const int lane = threadIdx.x & 31;
  const long long row = (long long)blockIdx.x * 8 + (threadIdx.x >> 5);
  const float* xr = x + row * CD;
  float s = 0.f, ss = 0.f;
  for (int j = lane; j < CD; j += 32) { float v = xr[j]; s += v; ss += v * v; }
#pragma unroll
  for (int o = 16; o; o >>= 1) { s += __shfl_xor(s, o); ss += __shfl_xor(ss, o); }
  const float m   = s * (1.0f / CD);
  const float var = ss * (1.0f / CD) - m * m;
  const float inv = rsqrtf(var + 1e-5f);
  u16* orow = out + row * CD;
  for (int j = lane; j < CD; j += 32)
    orow[j] = f2bf((xr[j] - m) * inv * w[j] + b[j]);
}

// wave-per-row causal softmax over T=1024; input f32 scores, output bf16 P.
// Covers H*T rows of one batch element (S/P laid out (H,T,T)).
__global__ __launch_bounds__(256)
void softmax_causal(const float* __restrict__ S, u16* __restrict__ P) {
  const int lane = threadIdx.x & 31;
  const long long rowId = (long long)blockIdx.x * 8 + (threadIdx.x >> 5);
  const int i = (int)(rowId & (CT - 1));
  const float* sr = S + rowId * CT;
  u16* pr = P + rowId * CT;
  const float scale = 0.04419417382415922f;   // 1/sqrt(512)
  float mx = -3.4e38f;
  for (int j = lane; j <= i; j += 32) mx = fmaxf(mx, sr[j] * scale);
#pragma unroll
  for (int o = 16; o; o >>= 1) mx = fmaxf(mx, __shfl_xor(mx, o));
  float sum = 0.f;
  for (int j = lane; j <= i; j += 32) sum += expf(sr[j] * scale - mx);
#pragma unroll
  for (int o = 16; o; o >>= 1) sum += __shfl_xor(sum, o);
  const float inv = 1.0f / sum;
  for (int j = lane; j < CT; j += 32)
    pr[j] = (j <= i) ? f2bf(expf(sr[j] * scale - mx) * inv) : (u16)0;
}

// V (B,T,H*D) bf16  ->  Vt (B,H,D,T) bf16, 32x32 LDS tiles
__global__ __launch_bounds__(256)
void transpose_v(const u16* __restrict__ v, u16* __restrict__ vt) {
  __shared__ u16 tile[32][33];
  const int bh = blockIdx.z;
  const int b  = bh >> 3, h = bh & 7;
  const int t0 = blockIdx.x * 32, d0 = blockIdx.y * 32;
  const int tx = threadIdx.x & 31, ty = threadIdx.x >> 5;   // 32 x 8
#pragma unroll
  for (int i = 0; i < 32; i += 8) {
    const int t = t0 + ty + i, d = d0 + tx;
    tile[ty + i][tx] = v[((long long)b * CT + t) * CHD + h * CD + d];
  }
  __syncthreads();
#pragma unroll
  for (int i = 0; i < 32; i += 8) {
    const int d = d0 + ty + i, t = t0 + tx;
    vt[((long long)bh * CD + d) * CT + t] = tile[tx][ty + i];
  }
}

// ---------------------------------------------------------------------------
// Host-side orchestration
// ---------------------------------------------------------------------------
extern "C" void kernel_launch(void* const* d_in, const int* in_sizes, int n_in,
                              void* d_out, int out_size, void* d_ws, size_t ws_size,
                              hipStream_t stream) {
  (void)in_sizes; (void)n_in; (void)out_size; (void)ws_size;

  const int*   idx    = (const int*)  d_in[0];
  const float* tok    = (const float*)d_in[1];
  const float* pos    = (const float*)d_in[2];
  const float* ln1w   = (const float*)d_in[3];
  const float* ln1b   = (const float*)d_in[4];
  const float* Wq     = (const float*)d_in[5];
  const float* bq     = (const float*)d_in[6];
  const float* Wk     = (const float*)d_in[7];
  const float* bk     = (const float*)d_in[8];
  const float* Wv     = (const float*)d_in[9];
  const float* bv     = (const float*)d_in[10];
  const float* Wp     = (const float*)d_in[11];
  const float* bp     = (const float*)d_in[12];
  const float* ln2w   = (const float*)d_in[13];
  const float* ln2b   = (const float*)d_in[14];
  const float* W1     = (const float*)d_in[15];
  const float* b1     = (const float*)d_in[16];
  const float* W2     = (const float*)d_in[17];
  const float* b2     = (const float*)d_in[18];
  const float* lnfw   = (const float*)d_in[19];
  const float* lnfb   = (const float*)d_in[20];
  const float* Whead  = (const float*)d_in[21];

  // workspace layout (bytes, all 256-aligned by construction)
  char* ws = (char*)d_ws;
  long long off = 0;
  float* xRes = (float*)(ws + off); off += (long long)CM * CD * 4;     // residual stream
  u16*   hxB  = (u16*)  (ws + off); off += (long long)CM * CD * 2;     // LN outputs (bf16)
  u16*   qB   = (u16*)  (ws + off); off += (long long)CM * CHD * 2;
  u16*   kB   = (u16*)  (ws + off); off += (long long)CM * CHD * 2;
  u16*   vB   = (u16*)  (ws + off); off += (long long)CM * CHD * 2;
  u16*   vtB  = (u16*)  (ws + off); off += (long long)CM * CHD * 2;    // (B,H,D,T)
  u16*   yB   = (u16*)  (ws + off); off += (long long)CM * CHD * 2;    // attn out
  u16*   mB   = (u16*)  (ws + off); off += (long long)CM * CFF * 2;    // MLP hidden
  float* Sb   = (float*)(ws + off); off += (long long)CH * CT * CT * 4;// scores, one b
  u16*   Pb   = (u16*)  (ws + off); off += (long long)CH * CT * CT * 2;// probs, one b
  u16*   wqB  = (u16*)  (ws + off); off += (long long)CHD * CD * 2;
  u16*   wkB  = (u16*)  (ws + off); off += (long long)CHD * CD * 2;
  u16*   wvB  = (u16*)  (ws + off); off += (long long)CHD * CD * 2;
  u16*   wpB  = (u16*)  (ws + off); off += (long long)CD * CHD * 2;
  u16*   w1B  = (u16*)  (ws + off); off += (long long)CFF * CD * 2;
  u16*   w2B  = (u16*)  (ws + off); off += (long long)CD * CFF * 2;
  u16*   whB  = (u16*)  (ws + off); off += (long long)CV * CD * 2;

  const dim3 blk(256);

  // x = tok_emb[idx] + pos_emb
  embed_kernel<<<(CM * CD) / 256, blk, 0, stream>>>(idx, tok, pos, xRes);

  // head weights to bf16 (once)
  cvt_f32_bf16<<<(CV * CD) / 256, blk, 0, stream>>>(Whead, whB, CV * CD);

  for (int l = 0; l < CL; ++l) {
    // per-layer weight conversion fp32 -> bf16
    cvt_f32_bf16<<<(CHD * CD) / 256, blk, 0, stream>>>(Wq + (long long)l * CHD * CD, wqB, CHD * CD);
    cvt_f32_bf16<<<(CHD * CD) / 256, blk, 0, stream>>>(Wk + (long long)l * CHD * CD, wkB, CHD * CD);
    cvt_f32_bf16<<<(CHD * CD) / 256, blk, 0, stream>>>(Wv + (long long)l * CHD * CD, wvB, CHD * CD);
    cvt_f32_bf16<<<(CD * CHD) / 256, blk, 0, stream>>>(Wp + (long long)l * CD * CHD, wpB, CD * CHD);
    cvt_f32_bf16<<<(CFF * CD) / 256, blk, 0, stream>>>(W1 + (long long)l * CFF * CD, w1B, CFF * CD);
    cvt_f32_bf16<<<(CD * CFF) / 256, blk, 0, stream>>>(W2 + (long long)l * CD * CFF, w2B, CD * CFF);

    // hx = LN1(x)  (bf16)
    ln_rows<<<CM / 8, blk, 0, stream>>>(xRes, ln1w + l * CD, ln1b + l * CD, hxB);

    // q/k/v = hx @ W^T + b   (M=8192, N=4096, K=512)  -> bf16
    gemm_bf16_wmma<true, true, false, false><<<dim3(CHD / 128, CM / 128, 1), blk, 0, stream>>>(
        hxB, 0, CD, wqB, 0, CD, qB, 0, CHD, bq + (long long)l * CHD, nullptr, 0, 0, CD);
    gemm_bf16_wmma<true, true, false, false><<<dim3(CHD / 128, CM / 128, 1), blk, 0, stream>>>(
        hxB, 0, CD, wkB, 0, CD, kB, 0, CHD, bk + (long long)l * CHD, nullptr, 0, 0, CD);
    gemm_bf16_wmma<true, true, false, false><<<dim3(CHD / 128, CM / 128, 1), blk, 0, stream>>>(
        hxB, 0, CD, wvB, 0, CD, vB, 0, CHD, bv + (long long)l * CHD, nullptr, 0, 0, CD);

    // Vt (B,H,D,T)
    transpose_v<<<dim3(CT / 32, CD / 32, CB * CH), blk, 0, stream>>>(vB, vtB);

    for (int b = 0; b < CB; ++b) {
      const long long qoff = (long long)b * CT * CHD;
      // S[h] = Q_bh (T x D) @ K_bh^T   (z = head, stride 512 columns)
      gemm_bf16_wmma<false, false, false, false><<<dim3(CT / 128, CT / 128, CH), blk, 0, stream>>>(
          qB + qoff, CD, CHD, kB + qoff, CD, CHD,
          Sb, (long long)CT * CT, CT, nullptr, nullptr, 0, 0, CD);
      // P = softmax(mask(S * scale))
      softmax_causal<<<(CH * CT) / 8, blk, 0, stream>>>(Sb, Pb);
      // Y_bh = P_bh (T x T) @ Vt_bh^T (Vt_bh is D x T)
      gemm_bf16_wmma<true, false, false, false><<<dim3(CD / 128, CT / 128, CH), blk, 0, stream>>>(
          Pb, (long long)CT * CT, CT,
          vtB + (long long)b * CH * CD * CT, (long long)CD * CT, CT,
          yB + qoff, CD, CHD, nullptr, nullptr, 0, 0, CT);
    }

    // x = x + y @ Wp^T + bp   (N=512, K=4096, residual, f32 out)
    gemm_bf16_wmma<false, true, true, false><<<dim3(CD / 128, CM / 128, 1), blk, 0, stream>>>(
        yB, 0, CHD, wpB, 0, CHD, xRes, 0, CD, bp + (long long)l * CD, xRes, 0, CD, CHD);

    // h2 = LN2(x)
    ln_rows<<<CM / 8, blk, 0, stream>>>(xRes, ln2w + l * CD, ln2b + l * CD, hxB);

    // m = gelu(h2 @ W1^T + b1)   (N=2048, K=512) -> bf16
    gemm_bf16_wmma<true, true, false, true><<<dim3(CFF / 128, CM / 128, 1), blk, 0, stream>>>(
        hxB, 0, CD, w1B, 0, CD, mB, 0, CFF, b1 + (long long)l * CFF, nullptr, 0, 0, CD);

    // x = x + m @ W2^T + b2   (N=512, K=2048, residual, f32 out)
    gemm_bf16_wmma<false, true, true, false><<<dim3(CD / 128, CM / 128, 1), blk, 0, stream>>>(
        mB, 0, CFF, w2B, 0, CFF, xRes, 0, CD, b2 + (long long)l * CD, xRes, 0, CD, CFF);
  }

  // xf = LNf(x)
  ln_rows<<<CM / 8, blk, 0, stream>>>(xRes, lnfw, lnfb, hxB);

  // logits = xf @ Whead^T   (M=8192, N=8192, K=512) -> f32 d_out
  gemm_bf16_wmma<false, false, false, false><<<dim3(CV / 128, CM / 128, 1), blk, 0, stream>>>(
      hxB, 0, CD, whB, 0, CD, (float*)d_out, 0, CV, nullptr, nullptr, 0, 0, CD);
}
